// IRNN_65223373357215
// MI455X (gfx1250) — matrix operand order repaired and checked by reading, not compiled, not run on hardware
//
#include <hip/hip_runtime.h>
#include <hip/hip_bf16.h>

// ---------------------------------------------------------------------------
// Fully-fused IRNN for MI455X (gfx1250, wave32, WMMA, TDM, 320KB LDS/WGP).
//
// One persistent kernel, 16 workgroups x 512 threads (16 wave32 waves).
// WG g owns batch rows [16g, 16g+16). Per timestep t:
//   u = emb[x_ids[b,t]] @ U^T + W_b   (16x256, K=128; 4 WMMAs/wave)
//   h = relu(h @ W^T + u)             (16x256, K=256; 8 WMMAs/wave)
//
// CDNA5-specific paths:
//  * v_wmma_f32_16x16x32_f16 (f16 in, f32 accumulate), 12 per wave per step
//  * TDM gather: one tensor_load_to_lds per step DMAs the 16-row embedding
//    gather (D# gather mode, 16-bit row indices, LDS padding 128dw+4dw ->
//    528B bank-conflict-free row pitch); drained by s_wait_tensorcnt
//  * loop-invariant W/U B-fragments pinned in VGPRs (96 regs)
//  * double-buffered h state -> ONE barrier per timestep instead of two
//  * bias pre-splatted in a register octet, fed as C of the first WMMA
// ---------------------------------------------------------------------------

#define B_ 256
#define T_ 512
#define V_ 32000
#define E_ 128
#define H_ 256
#define C_ 10

// padded LDS row strides: 528B / 272B row pitch
#define SW 264    // W rows, f16 (256 + 8)
#define SU 136    // U rows, f16 (128 + 8)
#define SXF 132   // x rows, f32 (128 + 4)  -- TDM pads 4 dwords per 128
#define SH 264    // h rows, f16 (256 + 8)

#define SMEM_BYTES ((256*SW + 256*SU + 2*16*SH) * 2 + (2*16*SXF + 256) * 4)

typedef __attribute__((ext_vector_type(16))) _Float16 v16h;
typedef __attribute__((ext_vector_type(8)))  _Float16 v8h;
typedef __attribute__((ext_vector_type(8)))  float    v8f;
typedef __attribute__((ext_vector_type(4)))  unsigned v4u;
typedef __attribute__((ext_vector_type(8)))  unsigned v8u;

// A-matrix 16x32 f16 fragment (ISA 7.12.2): lane L holds row m = L&15;
// lanes 0-15 cover K = [0,8)+[16,24), lanes 16-31 cover K = [8,16)+[24,32).
__device__ __forceinline__ v16h load_frag_a(const _Float16* row, int kbase, int g) {
    const v8h lo = *(const v8h*)(row + kbase + g * 8);
    const v8h hi = *(const v8h*)(row + kbase + 16 + g * 8);
    return __builtin_shufflevector(lo, hi, 0,1,2,3,4,5,6,7,8,9,10,11,12,13,14,15);
}

// Same fragment from f32 LDS rows: two 32B loads + v_cvt_pkrtz (VALU
// co-executes with the XDL/WMMA pipe).
__device__ __forceinline__ v16h load_frag_a_f32(const float* row, int kbase, int g) {
    const v8f lo = *(const v8f*)(row + kbase + g * 8);
    const v8f hi = *(const v8f*)(row + kbase + 16 + g * 8);
    v16h a;
    #pragma unroll
    for (int i = 0; i < 4; ++i) {
        auto p = __builtin_amdgcn_cvt_pkrtz(lo[2*i], lo[2*i+1]);
        a[2*i]   = p[0]; a[2*i+1]   = p[1];
        auto q = __builtin_amdgcn_cvt_pkrtz(hi[2*i], hi[2*i+1]);
        a[8+2*i] = q[0]; a[8+2*i+1] = q[1];
    }
    return a;
}

// B-matrix 32x16 f16 fragment: lane L holds column n = L&15; lanes 0-15 hold
// K = [0,8)+[16,24), lanes 16-31 hold K = [8,16)+[24,32) (matches A).
__device__ __forceinline__ v16h load_frag_b(const _Float16* row, int kbase, int g) {
    const v8h lo = *(const v8h*)(row + kbase + g * 8);
    const v8h hi = *(const v8h*)(row + kbase + 16 + g * 8);
    return __builtin_shufflevector(lo, hi, 0,1,2,3,4,5,6,7,8,9,10,11,12,13,14,15);
}

// ---------------------------------------------------------------------------
// TDM gather: load 16 rows of emb (row indices in g2/g3, 16-bit each) as a
// 16x128 f32 tile into LDS at lds_addr, padding 4 dwords per 128 dwords
// (row pitch 132 dwords = 528B). D# per ISA 08 §8.3-8.7.
// ---------------------------------------------------------------------------
__device__ __forceinline__ void tdm_gather_x_tile(const float* base, unsigned lds_addr,
                                                  v4u g2, v4u g3) {
    const unsigned long long ga = (unsigned long long)base;
    v4u g0;
    g0[0] = 0x80000001u;                           // count=1 | gather_mode=1 (16b idx)
    g0[1] = lds_addr;                              // lds_addr [63:32]
    g0[2] = (unsigned)(ga & 0xFFFFFFFFu);          // global_addr [95:64]
    g0[3] = (unsigned)((ga >> 32) & 0x01FFFFFFu)   // global_addr [120:96]
          | (2u << 30);                            // type=2 ("image")
    v8u g1;
    g1[0] = (2u << 16)                             // data_size: 4 bytes
          | (1u << 20)                             // pad_enable
          | (6u << 22)                             // pad_interval: 128 dwords
          | (3u << 25);                            // pad_amount: 4 dwords
    g1[1] = (128u << 16);                          // tensor_dim0 = 128 (E)
    g1[2] = ((unsigned)V_ << 16);                  // tensor_dim1 = V (rows, OOB)
    g1[3] = (128u << 16);                          // tile_dim0 = 128
    g1[4] = 16u;                                   // tile_dim1 = 16 valid indices
    g1[5] = 128u;                                  // tensor_dim0_stride = 128
    g1[6] = 0u;
    g1[7] = 0u;
    asm volatile("tensor_load_to_lds %0, %1, %2, %3"
                 :: "s"(g0), "s"(g1), "s"(g2), "s"(g3)
                 : "memory");
}

// Pack the 16 per-row embedding ids (held in lanes 0..15) into D# groups 2/3.
__device__ __forceinline__ void pack_gather_indices(int id, v4u* g2, v4u* g3) {
    unsigned d[8];
    #pragma unroll
    for (int j = 0; j < 8; ++j) {
        unsigned lo = (unsigned)__builtin_amdgcn_readlane(id, 2 * j);
        unsigned hi = (unsigned)__builtin_amdgcn_readlane(id, 2 * j + 1);
        d[j] = (lo & 0xFFFFu) | (hi << 16);
    }
    (*g2)[0] = d[0]; (*g2)[1] = d[1]; (*g2)[2] = d[2]; (*g2)[3] = d[3];
    (*g3)[0] = d[4]; (*g3)[1] = d[5]; (*g3)[2] = d[6]; (*g3)[3] = d[7];
}

__global__ __launch_bounds__(512, 1)
void irnn_fused_kernel(const int* __restrict__ x_ids,
                       const float* __restrict__ emb,
                       const float* __restrict__ U_w,
                       const float* __restrict__ W_w,
                       const float* __restrict__ W_b,
                       const float* __restrict__ ro_w,
                       const float* __restrict__ ro_b,
                       float* __restrict__ out) {
    extern __shared__ char smem_raw[];
    _Float16* Ws  = (_Float16*)smem_raw;           // [256][SW]   W as f16 (B[k,n]=W[n,k])
    _Float16* Us  = Ws + 256 * SW;                 // [256][SU]   U as f16
    _Float16* Hs  = Us + 256 * SU;                 // [2][16][SH] h state, double-buffered
    float*    Xsf = (float*)(Hs + 2 * 16 * SH);    // [2][16][SXF] TDM-landed x (f32)
    float*    Bs  = Xsf + 2 * 16 * SXF;            // [256]       bias

    const int tid  = threadIdx.x;
    const int lane = tid & 31;                     // wave32
    const int wave = tid >> 5;                     // 0..15
    const int g    = lane >> 4;                    // lane group
    const int mrow = lane & 15;                    // A-frag / C row id
    const int ncol = wave * 16 + (lane & 15);      // this lane's output column n
    const int b0   = blockIdx.x * 16;              // batch stripe base

    // ---- TDM-stage x(t=0); stage W (f16), U (f16), bias; zero h[0] --------
    if (wave == 0) {
        const int id = x_ids[(size_t)(b0 + (lane & 15)) * T_ + 0];
        v4u g2, g3;
        pack_gather_indices(id, &g2, &g3);
        tdm_gather_x_tile(emb, (unsigned)(uintptr_t)Xsf, g2, g3);
    }
    {   // W: thread handles one half-row (128 contiguous f32 -> f16)
        const int n = tid >> 1, half = tid & 1;
        const float4* src = (const float4*)(W_w + (size_t)n * H_ + half * 128);
        _Float16* dst = Ws + n * SW + half * 128;
        #pragma unroll
        for (int j = 0; j < 32; ++j) {
            float4 v = src[j];
            dst[4*j+0] = (_Float16)v.x; dst[4*j+1] = (_Float16)v.y;
            dst[4*j+2] = (_Float16)v.z; dst[4*j+3] = (_Float16)v.w;
        }
    }
    if (tid < 256) {                               // U: thread = one row
        const float4* src = (const float4*)(U_w + (size_t)tid * E_);
        _Float16* dst = Us + tid * SU;
        #pragma unroll
        for (int j = 0; j < 32; ++j) {
            float4 v = src[j];
            dst[4*j+0] = (_Float16)v.x; dst[4*j+1] = (_Float16)v.y;
            dst[4*j+2] = (_Float16)v.z; dst[4*j+3] = (_Float16)v.w;
        }
        Bs[tid] = W_b[tid];
    }
    for (int i = tid; i < 16 * SH; i += 512) Hs[i] = (_Float16)0.0f;  // h buffer 0

    if (wave == 0) __builtin_amdgcn_s_wait_tensorcnt(0);
    __syncthreads();

    // ---- Hoist loop-invariants into VGPRs ---------------------------------
    const _Float16* urow_p = Us + ncol * SU;
    const _Float16* wrow_p = Ws + ncol * SW;

    v16h wfrag[8], ufrag[4];
    #pragma unroll
    for (int kc = 0; kc < 8; ++kc) wfrag[kc] = load_frag_b(wrow_p, kc * 32, g);
    #pragma unroll
    for (int kc = 0; kc < 4; ++kc) ufrag[kc] = load_frag_b(urow_p, kc * 32, g);

    v8f acc0;                                      // bias splat, lives in regs
    {
        const float bias = Bs[ncol];
        #pragma unroll
        for (int i = 0; i < 8; ++i) acc0[i] = bias;
    }

    // ---- Recurrence over T timesteps (ONE barrier per step) ---------------
    for (int t = 0; t < T_; ++t) {
        const int cur = t & 1;
        const int nxt = cur ^ 1;
        const bool pf = (t + 1) < T_;

        // wave 0 kicks off the TDM gather of the t+1 x tile; the DMA engine
        // streams emb rows (hot in the 192MB L2) into LDS while all 16 waves
        // run this step's WMMAs.
        if (wave == 0 && pf) {
            const int id = x_ids[(size_t)(b0 + (lane & 15)) * T_ + (t + 1)];
            v4u g2, g3;
            pack_gather_indices(id, &g2, &g3);
            tdm_gather_x_tile(emb, (unsigned)(uintptr_t)(Xsf + nxt * (16 * SXF)),
                              g2, g3);
        }

        // u-part: x_tile[16x128] @ U^T; first WMMA consumes the bias octet
        const float* xrow_p = Xsf + cur * (16 * SXF) + mrow * SXF;
        v8f acc = __builtin_amdgcn_wmma_f32_16x16x32_f16(
            false, load_frag_a_f32(xrow_p, 0, g), false, ufrag[0],
            (short)0, acc0, false, false);
        #pragma unroll
        for (int kc = 1; kc < 4; ++kc) {
            v16h a = load_frag_a_f32(xrow_p, kc * 32, g);
            acc = __builtin_amdgcn_wmma_f32_16x16x32_f16(
                false, a, false, ufrag[kc], (short)0, acc, false, false);
        }

        // h-part: h[16x256] @ W^T  (the serial dependency chain)
        const _Float16* hrow_p = Hs + cur * (16 * SH) + mrow * SH;
        #pragma unroll
        for (int kc = 0; kc < 8; ++kc) {
            v16h a = load_frag_a(hrow_p, kc * 32, g);
            acc = __builtin_amdgcn_wmma_f32_16x16x32_f16(
                false, a, false, wfrag[kc], (short)0, acc, false, false);
        }

        // ReLU
        #pragma unroll
        for (int i = 0; i < 8; ++i) acc[i] = fmaxf(acc[i], 0.0f);

        // write new h into the OTHER buffer (C-layout: reg r -> row r + 8*g)
        _Float16* hdst = Hs + nxt * (16 * SH);
        #pragma unroll
        for (int r = 0; r < 8; ++r)
            hdst[(r + 8 * g) * SH + ncol] = (_Float16)acc[r];

        // drain this step's TDM gather, then one barrier publishes both the
        // new h buffer and the new x tile to all waves
        if (wave == 0 && pf) __builtin_amdgcn_s_wait_tensorcnt(0);
        __syncthreads();
    }

    // ---- Readout: out[b0+m, c] = h[m,:] . ro_w[c,:] + ro_b[c] -------------
    const _Float16* hfin = Hs + (T_ & 1) * (16 * SH);  // final h buffer
    if (tid < 16 * C_) {
        const int m = tid / C_, c = tid % C_;
        float s = ro_b[c];
        const float*    wr = ro_w + (size_t)c * H_;
        const _Float16* hr = hfin + m * SH;
        #pragma unroll 8
        for (int k = 0; k < H_; ++k) s += (float)hr[k] * wr[k];
        out[(size_t)(b0 + m) * C_ + c] = s;
    }
}

extern "C" void kernel_launch(void* const* d_in, const int* in_sizes, int n_in,
                              void* d_out, int out_size, void* d_ws, size_t ws_size,
                              hipStream_t stream) {
    (void)in_sizes; (void)n_in; (void)out_size; (void)d_ws; (void)ws_size;
    const int*   x_ids = (const int*)  d_in[0];
    const float* emb   = (const float*)d_in[1];
    const float* U_w   = (const float*)d_in[2];
    const float* W_w   = (const float*)d_in[3];
    const float* W_b   = (const float*)d_in[4];
    const float* ro_w  = (const float*)d_in[5];
    const float* ro_b  = (const float*)d_in[6];
    float* out = (float*)d_out;

    dim3 grid(B_ / 16);   // 16 persistent workgroups, one per WGP
    dim3 block(512);      // 16 wave32 waves
    irnn_fused_kernel<<<grid, block, SMEM_BYTES, stream>>>(
        x_ids, emb, U_w, W_w, W_b, ro_w, ro_b, out);
}
